// Gate_3401614099128
// MI455X (gfx1250) — compile-verified, matrix operand order checked by reading, not run on hardware
//
#include <hip/hip_runtime.h>
#include <hip/hip_bf16.h>

// ---------------------------------------------------------------------------
// MoE gate (router): scores = x @ W^T, softmax(64), top-2.
//   x: [N=16384, D=4096] f32   W: [E=64, D=4096] f32
// Out (flat, in order): topk_vals [N*2] f32 | topk_idx [N*2] i32 | probs [N*64] f32
//
// MI455X plan: HBM-bound on the single 256MB read of x (~11us @ 23.3TB/s).
// Arithmetic intensity 32 FLOP/byte -> needs ~780 TFLOP/s to stay BW-bound,
// which only the matrix pipe delivers: v_cvt_pk_bf16_f32 on the fly and
// v_wmma_f32_16x16x32_bf16 (16384 FLOP/instr; 524288 instrs total -> cheap).
// W (1MB) is pre-swizzled once into the exact WMMA B lane layout and stays
// L2-resident (192MB L2), so B fetches are contiguous 32B/lane L2 hits.
// LDS A tile is double-buffered (128-wide K chunk, one barrier per chunk);
// the pipeline is manually unrolled x2 so buffer parity is compile-time
// constant -> no cndmask address selects, clean ds_store_b128 codegen.
// ---------------------------------------------------------------------------

#define MODEL_DIM   4096
#define NUM_EXPERTS 64
#define TOPK        2
#define ROUTE_SCALE 1.0f

#define KCHUNK      128                    // K floats staged per iteration
#define NCHUNK      (MODEL_DIM / KCHUNK)   // 32
#define LDS_STRIDE  136                    // bf16 elems per row (pad: banks 4m%64)

typedef __attribute__((ext_vector_type(16))) __bf16 v16bf;
typedef __attribute__((ext_vector_type(2)))  __bf16 bf16x2;
typedef __attribute__((ext_vector_type(8)))  float  v8f;

union ABFrag { uint4 q[2]; v16bf v; };

// Pack two f32 into one dword of bf16x2 (RNE) -> v_cvt_pk_bf16_f32.
static __device__ __forceinline__ unsigned int pack_bf16x2(float a, float b) {
#if __has_builtin(__builtin_amdgcn_cvt_pk_bf16_f32)
    auto v = __builtin_amdgcn_cvt_pk_bf16_f32(a, b);
    return __builtin_bit_cast(unsigned int, v);
#else
    bf16x2 p = { (__bf16)a, (__bf16)b };
    return __builtin_bit_cast(unsigned int, p);
#endif
}

static __device__ __forceinline__ unsigned short to_bf16_u16(float a) {
    __bf16 h = (__bf16)a;
    return __builtin_bit_cast(unsigned short, h);
}

// Load 16 consecutive f32 (64B) of this thread's row slice.
static __device__ __forceinline__ void load_x16(const float* p, float4 r[4]) {
    r[0] = *(const float4*)(p + 0);
    r[1] = *(const float4*)(p + 4);
    r[2] = *(const float4*)(p + 8);
    r[3] = *(const float4*)(p + 12);
}

// Convert 16 f32 -> 16 bf16 and store as two contiguous b128 LDS stores.
static __device__ __forceinline__ void store_bf16x16(unsigned short* st,
                                                     const float4 r[4]) {
    uint4 p0 = { pack_bf16x2(r[0].x, r[0].y), pack_bf16x2(r[0].z, r[0].w),
                 pack_bf16x2(r[1].x, r[1].y), pack_bf16x2(r[1].z, r[1].w) };
    uint4 p1 = { pack_bf16x2(r[2].x, r[2].y), pack_bf16x2(r[2].z, r[2].w),
                 pack_bf16x2(r[3].x, r[3].y), pack_bf16x2(r[3].z, r[3].w) };
    *(uint4*)st       = p0;
    *(uint4*)(st + 8) = p1;
}

// Four K=32 WMMA steps over one staged 128-wide chunk.
static __device__ __forceinline__ void wmma_chunk(v8f& acc,
                                                  const unsigned short* fr,
                                                  const unsigned short*& bp) {
    #pragma unroll
    for (int sub = 0; sub < KCHUNK; sub += 32) {
        ABFrag a, b;
        a.q[0] = *(const uint4*)(fr + sub);
        a.q[1] = *(const uint4*)(fr + sub + 16);
        b.q[0] = *(const uint4*)(bp);
        b.q[1] = *(const uint4*)(bp + 8);
        bp += 32 * 16;
        acc = __builtin_amdgcn_wmma_f32_16x16x32_bf16(
            false, a.v, false, b.v, (short)0, acc, false, false);
    }
}

// ---------------------------------------------------------------------------
// Kernel 1: convert W f32 -> bf16 and swizzle into WMMA B-operand order.
// 16-bit operand layout (ISA 7.12.2, 16x32): lane = n16 + 16*((kk>>3)&1),
// element j = (kk&7) + 8*(kk>=16).  Stored as [n_tile][k_chunk32][lane][16].
// ---------------------------------------------------------------------------
__global__ void moe_gate_wswz(const float* __restrict__ w,
                              unsigned short* __restrict__ wsw) {
    int tid = blockIdx.x * blockDim.x + threadIdx.x;   // 0 .. 64*4096-1
    int n = tid >> 12;          // expert
    int k = tid & (MODEL_DIM - 1);
    int n_tile = n >> 4;
    int n16    = n & 15;
    int chunk  = k >> 5;        // 32-wide K chunk
    int kk     = k & 31;
    int lane   = n16 + (((kk >> 3) & 1) << 4);
    int j      = (kk & 7) + ((kk >> 4) << 3);
    int dst    = (((n_tile * (MODEL_DIM / 32) + chunk) * 32 + lane) << 4) + j;
    wsw[dst] = to_bf16_u16(w[tid]);
}

// ---------------------------------------------------------------------------
// Kernel 2: main gate. 128 threads = 4 waves. 16 rows/block, 16 experts/wave.
// ---------------------------------------------------------------------------
__global__ __launch_bounds__(128)
void moe_gate_main(const float* __restrict__ x,
                   const unsigned short* __restrict__ wsw,
                   float* __restrict__ topk_vals,
                   int*   __restrict__ topk_idx,
                   float* __restrict__ all_scores) {
    __shared__ unsigned short ldsA[2][16 * LDS_STRIDE];  // double-buffered A tile
    __shared__ float sc[16][65];                          // raw scores, padded
    __shared__ float s_max[16], s_rcp[16];

    const int t    = threadIdx.x;
    const int wave = t >> 5;        // expert tile 0..3
    const int lane = t & 31;
    const int rows0 = blockIdx.x * 16;

    // Staging role: thread t owns 16 consecutive f32 of row (t>>3) per chunk.
    const int sr   = t >> 3;
    const int scol = (t & 7) * 16;
    const float* xp = x + (size_t)(rows0 + sr) * MODEL_DIM + scol;
    unsigned short* const st0 = &ldsA[0][sr * LDS_STRIDE + scol];
    unsigned short* const st1 = &ldsA[1][sr * LDS_STRIDE + scol];

    // WMMA A fragment indices (16-bit A 16x32 layout).
    const int m    = lane & 15;
    const int koff = (lane >> 4) * 8;
    const unsigned short* const fr0 = &ldsA[0][m * LDS_STRIDE + koff];
    const unsigned short* const fr1 = &ldsA[1][m * LDS_STRIDE + koff];

    // Pre-swizzled B: per-wave tile base; streams forward 4KB per chunk.
    const unsigned short* bp =
        wsw + ((size_t)wave * (MODEL_DIM / 32) * 32 + lane) * 16;

    v8f acc = {};       // 16x16 f32 accumulator tile
    float4 r[4];

    // ---- prologue: stage chunk 0 into buffer 0 ----
    load_x16(xp, r);
    store_bf16x16(st0, r);
    __syncthreads();

    // ---- main pipeline, 2 chunks per iteration (compile-time parity) ----
    for (int c = 0; c < NCHUNK - 2; c += 2) {
        // even chunk c: compute buf0, stage chunk c+1 -> buf1
        load_x16(xp + (c + 1) * KCHUNK, r);
        wmma_chunk(acc, fr0, bp);
        store_bf16x16(st1, r);
        __syncthreads();

        // odd chunk c+1: compute buf1, stage chunk c+2 -> buf0
        load_x16(xp + (c + 2) * KCHUNK, r);
        wmma_chunk(acc, fr1, bp);
        store_bf16x16(st0, r);
        __syncthreads();
    }

    // ---- chunk 30 (buf0): stage final chunk 31 -> buf1 ----
    load_x16(xp + (NCHUNK - 1) * KCHUNK, r);
    wmma_chunk(acc, fr0, bp);
    store_bf16x16(st1, r);
    __syncthreads();

    // ---- chunk 31 (buf1): compute only ----
    wmma_chunk(acc, fr1, bp);

    // ---- scatter accumulator tile to LDS score board (ISA C/D layout) ----
    #pragma unroll
    for (int rr = 0; rr < 8; ++rr) {
        int M = rr + ((lane >> 4) << 3);
        sc[M][wave * 16 + (lane & 15)] = acc[rr];
    }
    __syncthreads();

    // ---- per-row softmax stats + top-2 (16 lanes, one row each) ----
    if (t < 16) {
        float mx = -__builtin_inff();
        #pragma unroll 8
        for (int e = 0; e < NUM_EXPERTS; ++e) mx = fmaxf(mx, sc[t][e]);
        float sum = 0.0f;
        float b1 = -__builtin_inff(), b2 = -__builtin_inff();
        int   i1 = 0, i2 = 0;
        for (int e = 0; e < NUM_EXPERTS; ++e) {
            float s = sc[t][e];
            sum += __expf(s - mx);
            if (s > b1)      { b2 = b1; i2 = i1; b1 = s; i1 = e; }
            else if (s > b2) { b2 = s; i2 = e; }
        }
        float rcp = 1.0f / sum;
        s_max[t] = mx;
        s_rcp[t] = rcp;
        int g = rows0 + t;
        topk_vals[g * TOPK + 0] = __expf(b1 - mx) * rcp * ROUTE_SCALE;
        topk_vals[g * TOPK + 1] = __expf(b2 - mx) * rcp * ROUTE_SCALE;
        topk_idx[g * TOPK + 0]  = i1;
        topk_idx[g * TOPK + 1]  = i2;
    }
    __syncthreads();

    // ---- write all 16x64 probabilities, fully coalesced ----
    float* probs = all_scores + (size_t)rows0 * NUM_EXPERTS;
    #pragma unroll
    for (int i = 0; i < 8; ++i) {
        int idx = i * 128 + t;              // consecutive threads -> consecutive addrs
        int rr = idx >> 6, e = idx & 63;
        probs[idx] = __expf(sc[rr][e] - s_max[rr]) * s_rcp[rr];
    }
}

extern "C" void kernel_launch(void* const* d_in, const int* in_sizes, int n_in,
                              void* d_out, int out_size, void* d_ws, size_t ws_size,
                              hipStream_t stream) {
    const float* x = (const float*)d_in[0];
    const float* w = (const float*)d_in[1];
    const int rows = in_sizes[0] / MODEL_DIM;   // 16384

    float* out_f      = (float*)d_out;
    float* topk_vals  = out_f;                                  // [rows*2] f32
    int*   topk_idx   = (int*)(out_f + (size_t)rows * TOPK);    // [rows*2] i32 (bit-exact)
    float* all_scores = out_f + (size_t)rows * TOPK * 2;        // [rows*64] f32

    unsigned short* wsw = (unsigned short*)d_ws;  // 512KB swizzled bf16 weights

    hipLaunchKernelGGL(moe_gate_wswz,
                       dim3((NUM_EXPERTS * MODEL_DIM) / 256), dim3(256), 0, stream,
                       w, wsw);
    hipLaunchKernelGGL(moe_gate_main,
                       dim3(rows / 16), dim3(128), 0, stream,
                       x, wsw, topk_vals, topk_idx, all_scores);
}
